// GRANDBlock_39273180955538
// MI455X (gfx1250) — matrix operand order repaired and checked by reading, not compiled, not run on hardware
//
#include <hip/hip_runtime.h>
#include <hip/hip_bf16.h>

// CDNA5 / gfx1250 WMMA types
typedef __attribute__((ext_vector_type(16))) __bf16 v16bf;
typedef __attribute__((ext_vector_type(8)))  __bf16 v8bf;
typedef __attribute__((ext_vector_type(8)))  float  v8f;

#define B_   8
#define H_   8
#define C_   1024
#define F_   256
#define D_   64
#define HD_  512
#define RHO_ 0.7f

#define WMMA_BF16(A, Bm, Cacc) \
  __builtin_amdgcn_wmma_f32_16x16x32_bf16(false, (A), false, (Bm), (short)0, (Cacc), false, false)

// concat two 16B bf16 vectors into one WMMA operand
__device__ __forceinline__ v16bf cat8(v8bf lo, v8bf hi) {
  return __builtin_shufflevector(lo, hi, 0, 1, 2, 3, 4, 5, 6, 7,
                                 8, 9, 10, 11, 12, 13, 14, 15);
}
// write 4 f32 -> 4 bf16 slots of a WMMA operand (packs to v_cvt_pk_bf16_f32)
#define PUT4(vec, off, f4)                                     \
  do {                                                         \
    (vec)[(off) + 0] = (__bf16)(f4).x;                         \
    (vec)[(off) + 1] = (__bf16)(f4).y;                         \
    (vec)[(off) + 2] = (__bf16)(f4).z;                         \
    (vec)[(off) + 3] = (__bf16)(f4).w;                         \
  } while (0)

// ---------------------------------------------------------------------------
// Kernel 1: fused QKV projection GEMM, LDS-staged weights.
// Grid: 64 row-blocks (8 waves x 16 rows) x 17 col-groups of 64
//       (groups 0-7 -> Wq, 8-15 -> Wk, 16 -> Wv).
// Per 32-wide K chunk: the block stages the 32x64 f32 weight chunk into LDS
// as bf16 in exact WMMA B-operand layout (vectorized b128 global reads, one
// conversion for all 8 waves), then each wave does 4 back-to-back WMMAs
// reusing one A operand.
// Q,K stored bf16 row-major [b][8192][64] (row i = c*8 + n/64 matches the
// reference's raw reshape); V stored transposed f32 [b][64][1024].
// ---------------------------------------------------------------------------
__global__ __launch_bounds__(256) void proj_qkv_kernel(
    const float* __restrict__ x,
    const float* __restrict__ Wq, const float* __restrict__ bq,
    const float* __restrict__ Wk, const float* __restrict__ bk,
    const float* __restrict__ Wv, const float* __restrict__ bv,
    __bf16* __restrict__ Qb, __bf16* __restrict__ Kb, float* __restrict__ Vt)
{
  __shared__ __align__(16) __bf16 WB[4][32][16];  // [n-tile][lane][K within 16]

  const int tid   = threadIdx.x;
  const int wave  = tid >> 5;
  const int lane  = tid & 31;
  const int halfl = lane >> 4;
  const int l16   = lane & 15;

  const int rb = blockIdx.x / 17;    // 64 row blocks
  const int g  = blockIdx.x % 17;    // 17 col groups of 64

  const float* W; const float* bp; int ncw; int base_col; int which;
  if (g < 8)       { W = Wq; bp = bq; ncw = HD_; base_col = g * 64;       which = 0; }
  else if (g < 16) { W = Wk; bp = bk; ncw = HD_; base_col = (g - 8) * 64; which = 1; }
  else             { W = Wv; bp = bv; ncw = D_;  base_col = 0;            which = 2; }

  const int tM = rb * 8 + wave;                     // 16-row tile index (0..511)
  const float* xbase = x + (size_t)(tM * 16 + l16) * F_;

  v8f acc[4] = {{}, {}, {}, {}};

  for (int kk = 0; kk < F_; kk += 32) {
    __syncthreads();     // previous chunk fully consumed by all waves

    // --- stage W[kk..kk+31][base_col..base_col+63] -> LDS bf16, B layout.
    // Consumer lane L=(khalf*16+col) wants 16 contiguous K values, i.e.
    // WB[n][khalf*16+col][j] = W[kk + khalf*16 + j][base_col + n*16 + col].
    // Global side: coalesced float4 (b128) reads; LDS side: 4 b16 scatters.
    #pragma unroll
    for (int rep = 0; rep < 2; ++rep) {
      const int idx = rep * 256 + tid;   // 0..511 float4 slots
      const int k   = idx >> 4;          // 0..31
      const int nn4 = (idx & 15) * 4;    // 0,4,...,60
      const float4 w4 =
          *(const float4*)&W[(size_t)(kk + k) * ncw + base_col + nn4];
      __bf16* dst = &WB[nn4 >> 4][(k >> 4) * 16 + (nn4 & 15)][k & 15];
      dst[0 * 16] = (__bf16)w4.x;        // consecutive cols = consecutive rows
      dst[1 * 16] = (__bf16)w4.y;
      dst[2 * 16] = (__bf16)w4.z;
      dst[3 * 16] = (__bf16)w4.w;
    }
    if (kk + 32 < F_)  // prefetch next weight chunk (global_prefetch_b8)
      __builtin_prefetch(&W[(size_t)(kk + 32 + (tid >> 3)) * ncw + base_col + ((tid & 7) << 3)], 0, 1);
    __syncthreads();

    // --- A operand: 16x32 x-tile (f32 -> bf16, explicit b128 loads).
    // Lane row = l16; lanes 0-15 hold K {0..15}, 16-31 K {16..31}.
    const float4* xr4a = (const float4*)(xbase + kk + halfl * 16);
    const float4* xr4b = (const float4*)(xbase + kk + halfl * 16 + 8);
    const float4 xa0 = xr4a[0], xa1 = xr4a[1];
    const float4 xa2 = xr4b[0], xa3 = xr4b[1];
    v16bf a;
    PUT4(a, 0, xa0); PUT4(a, 4, xa1); PUT4(a, 8, xa2); PUT4(a, 12, xa3);

    // --- 4 WMMAs, B tiles from LDS (2 x ds_load_b128 per operand)
    const v8bf* wb0 = (const v8bf*)&WB[0][lane][0];
    const v8bf* wb1 = (const v8bf*)&WB[1][lane][0];
    const v8bf* wb2 = (const v8bf*)&WB[2][lane][0];
    const v8bf* wb3 = (const v8bf*)&WB[3][lane][0];
    acc[0] = WMMA_BF16(a, cat8(wb0[0], wb0[1]), acc[0]);
    acc[1] = WMMA_BF16(a, cat8(wb1[0], wb1[1]), acc[1]);
    acc[2] = WMMA_BF16(a, cat8(wb2[0], wb2[1]), acc[2]);
    acc[3] = WMMA_BF16(a, cat8(wb3[0], wb3[1]), acc[3]);
  }

  // --- epilogue: bias + scatter-store in the reference reshape layout.
  // C/D layout: lane holds one column; VGPR r holds row M = r + 8*halfl.
  #pragma unroll
  for (int n = 0; n < 4; ++n) {
    const int ncol = base_col + n * 16 + l16;   // column within selected matrix
    const float bias = bp[ncol];
    #pragma unroll
    for (int r = 0; r < 8; ++r) {
      const int grow = tM * 16 + halfl * 8 + r; // = b*1024 + c
      const int b = grow >> 10, c = grow & 1023;
      const float val = acc[n][r] + bias;
      if (which == 0)
        Qb[((size_t)b * 8192 + c * 8 + (ncol >> 6)) * 64 + (ncol & 63)] = (__bf16)val;
      else if (which == 1)
        Kb[((size_t)b * 8192 + c * 8 + (ncol >> 6)) * 64 + (ncol & 63)] = (__bf16)val;
      else
        Vt[((size_t)b * 64 + ncol) * 1024 + c] = val;     // transposed V
    }
  }
}

// ---------------------------------------------------------------------------
// Kernel 2: fused masked-softmax attention.
// One block = one batch b and one 16-row query panel c0..c0+15 (64 KB score
// panel resident in LDS; CDNA5 WGP has 320 KB). For each head h:
//   A) 8 waves compute the 16x1024 score panel (Q.K^T / 8) via WMMA -> LDS,
//      Q A-operands hoisted and reused across all 8 key tiles per wave
//   B) cooperative row max / sumexp (softmax #1 stats), float4 sweeps
//   C) threshold mask (sm > RHO ? s : 0) in place, stats #2, att -> LDS
//   D) ALL 8 waves: out_acc += att[16x1024] @ V[1024x64]; the K=1024
//      reduction is split across wave pairs (0-3: kk<512, 4-7: kk>=512),
//      partials combined once through LDS after the head loop.
// Epilogue: out = out_acc/H - V[row]  (the "att_map - I" term).
// ---------------------------------------------------------------------------
__global__ __launch_bounds__(256) void attn_kernel(
    const __bf16* __restrict__ Qb, const __bf16* __restrict__ Kb,
    const float* __restrict__ Vt, float* __restrict__ out)
{
  __shared__ __align__(16) float S[16][1024];   // 64 KB score/att panel
  __shared__ float red[16][16];                 // cross-wave reduction scratch
  __shared__ float rstat[4][16];                // max1, sum1, max2, sum2

  const int b  = blockIdx.x >> 6;   // 8 batches
  const int ct = blockIdx.x & 63;   // 64 row panels of 16
  const int c0 = ct * 16;

  const int wave  = threadIdx.x >> 5;
  const int lane  = threadIdx.x & 31;
  const int halfl = lane >> 4;
  const int l16   = lane & 15;

  const int rr = threadIdx.x >> 4;  // reduction: row 0..15
  const int ss = threadIdx.x & 15;  // reduction: float4-group 0..15

  const __bf16* Qbase = Qb + (size_t)b * 8192 * 64;
  const __bf16* Kbase = Kb + (size_t)b * 8192 * 64;
  const float*  Vtb   = Vt + (size_t)b * 64 * 1024;

  float4* const Srow4 = (float4*)&S[rr][0];     // this thread's reduction row

  // Phase-D work split: output slab d0, K-range [k0, k0+512)
  const int d0 = (wave & 3) * 16;
  const int k0 = (wave >> 2) * 512;

  v8f oacc = {};                    // per-wave partial output accumulator

  for (int h = 0; h < H_; ++h) {
    // ---- Phase A: scores -> LDS. Wave w owns key columns [w*128, w*128+128).
    // Hoist the two Q A-operands (K-dim 64 = 2 x 32) out of the tile loop.
    const __bf16* qrow = Qbase + (size_t)(h * 1024 + c0 + l16) * 64 + halfl * 16;
    const v8bf* q8 = (const v8bf*)qrow;           // 16B-aligned
    const v16bf qa0 = cat8(q8[0], q8[1]);         // K 0..31 chunk
    const v16bf qa1 = cat8(q8[4], q8[5]);         // K 32..63 chunk
    #pragma unroll 4
    for (int t = 0; t < 8; ++t) {
      const int cc0 = wave * 128 + t * 16;
      const v8bf* k8 =
          (const v8bf*)(Kbase + (size_t)(h * 1024 + cc0 + l16) * 64 + halfl * 16);
      v8f sacc = {};
      sacc = WMMA_BF16(qa0, cat8(k8[0], k8[1]), sacc);
      sacc = WMMA_BF16(qa1, cat8(k8[4], k8[5]), sacc);
      #pragma unroll
      for (int r = 0; r < 8; ++r)
        S[halfl * 8 + r][cc0 + l16] = sacc[r] * 0.125f;   // 1/sqrt(D)
    }
    __syncthreads();

    // ---- Phase B: softmax #1 stats (max1, sum1) per row (float4 sweeps).
    float m = -3.0e38f;
    for (int j = ss; j < 256; j += 16) {
      const float4 v = Srow4[j];
      m = fmaxf(m, fmaxf(fmaxf(v.x, v.y), fmaxf(v.z, v.w)));
    }
    red[rr][ss] = m;
    __syncthreads();
    if (ss == 0) {
      float mm = -3.0e38f;
      #pragma unroll
      for (int j = 0; j < 16; ++j) mm = fmaxf(mm, red[rr][j]);
      rstat[0][rr] = mm;
    }
    __syncthreads();
    const float m1 = rstat[0][rr];
    float sum = 0.f;
    for (int j = ss; j < 256; j += 16) {
      const float4 v = Srow4[j];
      sum += __expf(v.x - m1) + __expf(v.y - m1) +
             __expf(v.z - m1) + __expf(v.w - m1);
    }
    red[rr][ss] = sum;
    __syncthreads();
    if (ss == 0) {
      float s2 = 0.f;
      #pragma unroll
      for (int j = 0; j < 16; ++j) s2 += red[rr][j];
      rstat[1][rr] = s2;
    }
    __syncthreads();

    // ---- Phase C: hard-threshold mask in place, then stats #2, then att.
    const float inv1 = 1.f / rstat[1][rr];
    for (int j = ss; j < 256; j += 16) {
      float4 v = Srow4[j];
      v.x = (__expf(v.x - m1) * inv1 > RHO_) ? v.x : 0.f;
      v.y = (__expf(v.y - m1) * inv1 > RHO_) ? v.y : 0.f;
      v.z = (__expf(v.z - m1) * inv1 > RHO_) ? v.z : 0.f;
      v.w = (__expf(v.w - m1) * inv1 > RHO_) ? v.w : 0.f;
      Srow4[j] = v;
    }
    __syncthreads();
    m = -3.0e38f;
    for (int j = ss; j < 256; j += 16) {
      const float4 v = Srow4[j];
      m = fmaxf(m, fmaxf(fmaxf(v.x, v.y), fmaxf(v.z, v.w)));
    }
    red[rr][ss] = m;
    __syncthreads();
    if (ss == 0) {
      float mm = -3.0e38f;
      #pragma unroll
      for (int j = 0; j < 16; ++j) mm = fmaxf(mm, red[rr][j]);
      rstat[2][rr] = mm;
    }
    __syncthreads();
    const float m2 = rstat[2][rr];
    sum = 0.f;
    for (int j = ss; j < 256; j += 16) {
      const float4 v = Srow4[j];
      sum += __expf(v.x - m2) + __expf(v.y - m2) +
             __expf(v.z - m2) + __expf(v.w - m2);
    }
    red[rr][ss] = sum;
    __syncthreads();
    if (ss == 0) {
      float s2 = 0.f;
      #pragma unroll
      for (int j = 0; j < 16; ++j) s2 += red[rr][j];
      rstat[3][rr] = s2;
    }
    __syncthreads();
    const float inv2 = 1.f / rstat[3][rr];
    for (int j = ss; j < 256; j += 16) {         // att row (per head)
      float4 v = Srow4[j];
      v.x = __expf(v.x - m2) * inv2;
      v.y = __expf(v.y - m2) * inv2;
      v.z = __expf(v.z - m2) * inv2;
      v.w = __expf(v.w - m2) * inv2;
      Srow4[j] = v;
    }
    __syncthreads();

    // ---- Phase D: out_acc += att @ V, all 8 waves (K split across pairs).
    #pragma unroll 2
    for (int kk = k0; kk < k0 + 512; kk += 32) {
      // A operand: att rows from LDS (f32 -> bf16), 2+2 x ds_load_b128
      const float4* sa = (const float4*)&S[l16][kk + halfl * 16];
      const float4* sb = (const float4*)&S[l16][kk + halfl * 16 + 8];
      const float4 a0 = sa[0], a1 = sa[1], a2 = sb[0], a3 = sb[1];
      v16bf a;
      PUT4(a, 0, a0); PUT4(a, 4, a1); PUT4(a, 8, a2); PUT4(a, 12, a3);
      // B operand: V^T rows (f32 -> bf16), 4 x global_load_b128
      const float4* vt4 =
          (const float4*)(Vtb + (size_t)(d0 + l16) * 1024 + kk + halfl * 16);
      const float4 w0 = vt4[0], w1 = vt4[1], w2 = vt4[2], w3 = vt4[3];
      v16bf bm;
      PUT4(bm, 0, w0); PUT4(bm, 4, w1); PUT4(bm, 8, w2); PUT4(bm, 12, w3);
      oacc = WMMA_BF16(a, bm, oacc);
    }
    __syncthreads();   // S reused next head
  }

  // ---- Combine the two K-half partials through LDS (S is dead now).
  if (wave >= 4) {
    float* dst = &S[0][0] + ((wave - 4) * 32 + lane) * 8;
    #pragma unroll
    for (int r = 0; r < 8; ++r) dst[r] = oacc[r];
  }
  __syncthreads();

  // ---- Epilogue: mean over heads, subtract identity*V, store.
  if (wave < 4) {
    const float* src = &S[0][0] + (wave * 32 + lane) * 8;
    const int d = wave * 16 + l16;
    #pragma unroll
    for (int r = 0; r < 8; ++r) {
      const int row = c0 + halfl * 8 + r;
      const float vI = Vtb[(size_t)d * 1024 + row];
      out[((size_t)b * 1024 + row) * 64 + d] =
          (oacc[r] + src[r]) * (1.0f / H_) - vI;
    }
  }
}

// ---------------------------------------------------------------------------
extern "C" void kernel_launch(void* const* d_in, const int* in_sizes, int n_in,
                              void* d_out, int out_size, void* d_ws, size_t ws_size,
                              hipStream_t stream) {
  const float* x  = (const float*)d_in[0];
  const float* Wq = (const float*)d_in[1];
  const float* bq = (const float*)d_in[2];
  const float* Wk = (const float*)d_in[3];
  const float* bk = (const float*)d_in[4];
  const float* Wv = (const float*)d_in[5];
  const float* bv = (const float*)d_in[6];
  float* out = (float*)d_out;

  // Workspace layout: Q bf16 (8 MB) | K bf16 (8 MB) | V^T f32 (2 MB)
  const size_t qk_elems = (size_t)B_ * 8192 * 64;
  __bf16* Qb = (__bf16*)d_ws;
  __bf16* Kb = (__bf16*)((char*)d_ws + qk_elems * 2);
  float*  Vt = (float*) ((char*)d_ws + qk_elems * 4);

  // 64 row-blocks x 17 col-groups
  proj_qkv_kernel<<<dim3(64 * 17), dim3(256), 0, stream>>>(
      x, Wq, bq, Wk, bk, Wv, bv, Qb, Kb, Vt);

  // 8 batches x 64 query panels
  attn_kernel<<<dim3(B_ * 64), dim3(256), 0, stream>>>(Qb, Kb, Vt, out);

  (void)in_sizes; (void)n_in; (void)out_size; (void)ws_size;
}
